// SwinTransformerBlock_22101901705473
// MI455X (gfx1250) — matrix operand order repaired and checked by reading, not compiled, not run on hardware
//
#include <hip/hip_runtime.h>
#include <hip/hip_bf16.h>

// ---------------------------------------------------------------------------
// Swin Transformer block, MI455X (gfx1250, wave32, WMMA).
// fp16 storage / fp32-accumulate WMMA for all GEMMs; fp32 LN/softmax/residual.
// GEMM tiles staged with GLOBAL_LOAD_ASYNC_TO_LDS_B128 (ASYNCcnt), double-
// buffered in LDS; WMMA fragments read back per ISA 7.12.2 layouts.
// ---------------------------------------------------------------------------

typedef __attribute__((ext_vector_type(16))) _Float16 v16h;
typedef __attribute__((ext_vector_type(8)))  _Float16 v8h;
typedef __attribute__((ext_vector_type(8)))  float    v8f;

#define WMMA_F16(a, b, c) \
  __builtin_amdgcn_wmma_f32_16x16x32_f16(false, (a), false, (b), (short)0, (c), false, false)

// Problem constants
constexpr int kB   = 8;
constexpr int kC   = 256;
constexpr int kSS  = 4;
constexpr int kNH  = 8;
constexpr int kHD  = 32;
constexpr int kNW  = 256;               // windows per image
constexpr int kT   = kB * kNW * 64;     // 131072 total tokens
constexpr float kQScale = 0.17677669529663687f;  // 32^-0.5

// Workspace layout (bytes), liveness-based reuse:
//   buf0 : xw16 (LN1 out) -> attn16 (attention out) -> m16 (LN2 out)
//   qkv  : q/k/v planes; dead after attention -> reused by h16
//   x1   : fp32 residual after first block;  wts : fp16 weight copies
constexpr size_t SZ_TC2   = (size_t)kT * kC * 2;             // 67,108,864
constexpr size_t OFF_BUF0 = 0;
constexpr size_t OFF_Q    = SZ_TC2;
constexpr size_t OFF_K    = 2 * SZ_TC2;
constexpr size_t OFF_V    = 3 * SZ_TC2;
constexpr size_t OFF_H16  = SZ_TC2;                          // reuse qkv region
constexpr size_t OFF_X1   = OFF_H16 + (size_t)kT * 1024 * 2; // 335,544,320
constexpr size_t OFF_WQKV = OFF_X1 + (size_t)kT * kC * 4;    // 469,762,048
constexpr size_t OFF_WPRJ = OFF_WQKV + (size_t)768 * 256 * 2;
constexpr size_t OFF_WFC1 = OFF_WPRJ + (size_t)256 * 256 * 2;
constexpr size_t OFF_WFC2 = OFF_WFC1 + (size_t)1024 * 256 * 2;

// ------------------------ CDNA5 async-to-LDS helpers -----------------------
// GVS mode: mem_addr = SGPR64(base) + VGPR32(byte offset); per-lane 16B -> LDS.
__device__ __forceinline__ void async_ld_b128(unsigned lds_off, unsigned gl_off,
                                              const void* base) {
  asm volatile("global_load_async_to_lds_b128 %0, %1, %2"
               :
               : "v"(lds_off), "v"(gl_off), "s"(base)
               : "memory");
}
__device__ __forceinline__ void wait_async0() {
  asm volatile("s_wait_asynccnt 0x0" ::: "memory");
}
__device__ __forceinline__ void wait_async6() {
  asm volatile("s_wait_asynccnt 0x6" ::: "memory");
}
__device__ __forceinline__ unsigned lds_addr(const void* p) {
  return (unsigned)(size_t)p;  // generic shared addr: [31:0] = LDS offset
}

// --------------------------- fragment loaders ------------------------------
// A fragment, 16x32 f16 (ISA 7.12.2): lane row = row0+(lane&15);
// elements 0..7 -> K = 8*(lane>>4)+0..7, elements 8..15 -> +16.
__device__ __forceinline__ v16h ld_afrag(const _Float16* base, int row0, int kbase,
                                         int stride, int lane) {
  const _Float16* p = base + (size_t)(row0 + (lane & 15)) * stride + kbase + ((lane >> 4) << 3);
  v8h lo = *(const v8h*)(p);
  v8h hi = *(const v8h*)(p + 16);
  v16h r;
#pragma unroll
  for (int e = 0; e < 8; ++e) { r[e] = lo[e]; r[e + 8] = hi[e]; }
  return r;
}

// B fragment, 32x16 f16: lane col = col0+(lane&15);
// elements 0..7 -> K = 8*(lane>>4)+0..7, elements 8..15 -> +16.
__device__ __forceinline__ v16h ld_bfrag(const _Float16* base, int col0, int kbase,
                                         int stride, int lane) {
  const _Float16* p = base + (size_t)(col0 + (lane & 15)) * stride + kbase + ((lane >> 4) << 3);
  v8h lo = *(const v8h*)(p);
  v8h hi = *(const v8h*)(p + 16);
  v16h r;
#pragma unroll
  for (int e = 0; e < 8; ++e) { r[e] = lo[e]; r[e + 8] = hi[e]; }
  return r;
}

// Windowed token -> (batch, src_h, src_w) pixel (cyclic shift by +SS, wrap 128)
__device__ __forceinline__ size_t token_src_pixel(int t) {
  int i = t & 63, win = t >> 6;
  int b = win >> 8, w = win & 255;
  int wy = w >> 4, wx = w & 15;
  int h = wy * 8 + (i >> 3), wc = wx * 8 + (i & 7);
  int hs = (h + kSS) & 127, wsr = (wc + kSS) & 127;
  return ((size_t)b * 16384 + (size_t)hs * 128 + wsr);
}

// --------------------------- small kernels ---------------------------------
__global__ __launch_bounds__(256) void cvt_f32_f16(const float* __restrict__ src,
                                                   _Float16* __restrict__ dst, int n) {
  int i = blockIdx.x * 256 + threadIdx.x;
  if (i < n) dst[i] = (_Float16)src[i];
}

// LN over C=256: one wave per token, 8 channels/lane.
// MODE==0: windowed order + shift (reads x), MODE==1: natural order (reads x1).
template <int MODE>
__global__ __launch_bounds__(256) void ln_kernel(const float* __restrict__ src,
                                                 const float* __restrict__ g,
                                                 const float* __restrict__ bb,
                                                 _Float16* __restrict__ dst) {
  const int t = blockIdx.x * 8 + (threadIdx.x >> 5);
  const int lane = threadIdx.x & 31;
  size_t base;
  if constexpr (MODE == 0) base = token_src_pixel(t) * kC;
  else                     base = (size_t)t * kC;
  float4 v0 = *(const float4*)(src + base + lane * 8);
  float4 v1 = *(const float4*)(src + base + lane * 8 + 4);
  float s  = v0.x + v0.y + v0.z + v0.w + v1.x + v1.y + v1.z + v1.w;
  float s2 = v0.x*v0.x + v0.y*v0.y + v0.z*v0.z + v0.w*v0.w +
             v1.x*v1.x + v1.y*v1.y + v1.z*v1.z + v1.w*v1.w;
#pragma unroll
  for (int m = 1; m < 32; m <<= 1) { s += __shfl_xor(s, m, 32); s2 += __shfl_xor(s2, m, 32); }
  const float mean = s * (1.0f / 256.0f);
  const float var  = s2 * (1.0f / 256.0f) - mean * mean;
  const float rstd = rsqrtf(var + 1e-5f);
  const int c0 = lane * 8;
  float vv[8] = {v0.x, v0.y, v0.z, v0.w, v1.x, v1.y, v1.z, v1.w};
  v8h o;
#pragma unroll
  for (int e = 0; e < 8; ++e)
    o[e] = (_Float16)((vv[e] - mean) * rstd * g[c0 + e] + bb[c0 + e]);
  *(v8h*)(dst + (size_t)t * kC + c0) = o;
}

// --------------------------- WMMA GEMM -------------------------------------
// C[m,n] = sum_k A[m,k] * W[n,k] (+bias, +epilogue).  8 waves, WG tile 128x256,
// wave tile 64x64 (4x4 accumulators).  K staged in 32-wide slices via
// GLOBAL_LOAD_ASYNC_TO_LDS_B128 into double-buffered LDS (ASYNCcnt pipelined).
enum { EPI_QKV = 0, EPI_PROJ = 1, EPI_FC1 = 2, EPI_FC2 = 3 };

template <int KDIM, int EPI>
__global__ __launch_bounds__(256) void gemm_wmma_kernel(
    const _Float16* __restrict__ A, const _Float16* __restrict__ Wt,
    const float* __restrict__ bias,
    _Float16* __restrict__ oq, _Float16* __restrict__ ok, _Float16* __restrict__ ov,
    const float* __restrict__ addsrc, float* __restrict__ o32) {
  constexpr int KSTEPS = KDIM / 32;
  constexpr int ASZ = 128 * 40;   // halves per stage
  constexpr int BSZ = 256 * 40;
  __shared__ __align__(16) _Float16 lsA[2 * ASZ];
  __shared__ __align__(16) _Float16 lsB[2 * BSZ];

  const int tid  = threadIdx.x;
  const int lane = tid & 31;
  const int wv   = tid >> 5;
  const int wm   = wv & 1;        // wave grid 2 (M) x 4 (N)
  const int wn   = wv >> 1;
  const size_t mblk = (size_t)blockIdx.x * 128;
  const size_t nblk = (size_t)blockIdx.y * 256;

  v8f acc[4][4];
#pragma unroll
  for (int mi = 0; mi < 4; ++mi)
#pragma unroll
    for (int ni = 0; ni < 4; ++ni) acc[mi][ni] = (v8f)0.0f;

  // Issue one K-slice (6 async b128 per thread: 2 for A-tile, 4 for B-tile).
  auto issue = [&](int ks, int buf) {
#pragma unroll
    for (int it = 0; it < 2; ++it) {
      int id = tid + it * 256;
      int row = id >> 2, ch = id & 3;
      unsigned go = (unsigned)((((mblk + row) * KDIM) + ks * 32 + ch * 8) * 2);
      async_ld_b128(lds_addr(&lsA[buf * ASZ + row * 40 + ch * 8]), go, A);
    }
#pragma unroll
    for (int it = 0; it < 4; ++it) {
      int id = tid + it * 256;
      int row = id >> 2, ch = id & 3;
      unsigned go = (unsigned)((((nblk + row) * KDIM) + ks * 32 + ch * 8) * 2);
      async_ld_b128(lds_addr(&lsB[buf * BSZ + row * 40 + ch * 8]), go, Wt);
    }
  };

  issue(0, 0);
#pragma unroll 1
  for (int ks = 0; ks < KSTEPS; ++ks) {
    if (ks + 1 < KSTEPS) {
      issue(ks + 1, (ks + 1) & 1);  // overlap next slice with this one's wait
      wait_async6();                 // in-order: slice ks complete
    } else {
      wait_async0();
    }
    __syncthreads();

    const _Float16* tA = lsA + (ks & 1) * ASZ;
    const _Float16* tB = lsB + (ks & 1) * BSZ;
    v16h af[4];
#pragma unroll
    for (int mi = 0; mi < 4; ++mi) af[mi] = ld_afrag(tA, wm * 64 + mi * 16, 0, 40, lane);
#pragma unroll
    for (int ni = 0; ni < 4; ++ni) {
      v16h bf = ld_bfrag(tB, wn * 64 + ni * 16, 0, 40, lane);
#pragma unroll
      for (int mi = 0; mi < 4; ++mi) acc[mi][ni] = WMMA_F16(af[mi], bf, acc[mi][ni]);
    }
    __syncthreads();  // all waves done reading before buffer parity reuse
  }

  // Epilogue.  D layout: VGPR r, lane l -> row r + 8*(l>>4), col l&15.
  const int rhalf = (lane >> 4) * 8;
  const int colL  = lane & 15;
#pragma unroll
  for (int mi = 0; mi < 4; ++mi) {
#pragma unroll
    for (int ni = 0; ni < 4; ++ni) {
      const int n = (int)nblk + wn * 64 + ni * 16 + colL;
      const float bn = bias[n];
      const int mbase = (int)mblk + wm * 64 + mi * 16 + rhalf;
#pragma unroll
      for (int r = 0; r < 8; ++r) {
        const int m = mbase + r;
        float y = acc[mi][ni][r] + bn;
        if constexpr (EPI == EPI_QKV) {
          // plane uniform per block: blockIdx.y = 0:q 1:k 2:v
          _Float16* o = (blockIdx.y == 0) ? oq : ((blockIdx.y == 1) ? ok : ov);
          if (blockIdx.y == 0) y *= kQScale;
          o[(size_t)m * kC + (n & 255)] = (_Float16)y;
        } else if constexpr (EPI == EPI_PROJ) {
          // window reverse + reverse cyclic shift + residual (f32)
          const size_t idx = token_src_pixel(m) * kC + n;
          o32[idx] = addsrc[idx] + y;
        } else if constexpr (EPI == EPI_FC1) {
          const float gl = 0.5f * y * (1.0f + erff(y * 0.70710678118654752f));
          oq[(size_t)m * 1024 + n] = (_Float16)gl;
        } else {  // EPI_FC2
          o32[(size_t)m * kC + n] = addsrc[(size_t)m * kC + n] + y;
        }
      }
    }
  }
}

// --------------------------- attention -------------------------------------
// One block per (window, head): 4 waves x 16 rows. S=Q.K^T (WMMA, K=HD=32),
// +rel-pos bias +shift mask, f32 softmax, P.V (WMMA, K=64 in 2 steps).
__global__ __launch_bounds__(128) void attn_kernel(
    const _Float16* __restrict__ q16, const _Float16* __restrict__ k16,
    const _Float16* __restrict__ v16, const float* __restrict__ rpb,
    _Float16* __restrict__ out16) {
  __shared__ __align__(16) _Float16 lsQ[64 * 40];
  __shared__ __align__(16) _Float16 lsK[64 * 40];
  __shared__ __align__(16) _Float16 lsVt[32 * 72];  // [dim][token]
  __shared__ __align__(16) _Float16 lsP[64 * 72];
  __shared__ float lsBias[225];

  const int head = blockIdx.x;
  const int win  = blockIdx.y;
  const int tid  = threadIdx.x, lane = tid & 31, wv = tid >> 5;

  if (tid < 225) lsBias[tid] = rpb[tid * kNH + head];

  const size_t tokbase = (size_t)win * 64 * kC + head * kHD;
#pragma unroll
  for (int it = 0; it < 2; ++it) {
    int id = tid + it * 128;        // 256 chunks of 8 halves per tensor
    int row = id >> 2, ch = id & 3;
    const size_t g = tokbase + (size_t)row * kC + ch * 8;
    *(uint4*)&lsQ[row * 40 + ch * 8] = *(const uint4*)(q16 + g);
    *(uint4*)&lsK[row * 40 + ch * 8] = *(const uint4*)(k16 + g);
    v8h vv = *(const v8h*)(v16 + g);
#pragma unroll
    for (int e = 0; e < 8; ++e) lsVt[(ch * 8 + e) * 72 + row] = vv[e];  // transpose
  }
  __syncthreads();

  const int r0 = wv * 16;
  v8f acc[4];
#pragma unroll
  for (int ni = 0; ni < 4; ++ni) acc[ni] = (v8f)0.0f;
  {
    v16h aq = ld_afrag(lsQ, r0, 0, 40, lane);
#pragma unroll
    for (int ni = 0; ni < 4; ++ni) {
      v16h bk = ld_bfrag(lsK, ni * 16, 0, 40, lane);
      acc[ni] = WMMA_F16(aq, bk, acc[ni]);
    }
  }

  // bias + shift mask + row softmax (rows live in one 16-lane half)
  const int rhalf = lane >> 4, colL = lane & 15;
  const int w = win & 255, wy = w >> 4, wx = w & 15;
  auto region = [](int p) { return (p < 120) ? 0 : ((p < 124) ? 1 : 2); };
#pragma unroll
  for (int r = 0; r < 8; ++r) {
    const int i = r0 + r + rhalf * 8;
    const int yi = i >> 3, xi = i & 7;
    const int ri = region(wy * 8 + yi) * 3 + region(wx * 8 + xi);
    float vals[4];
    float vmax = -1e30f;
#pragma unroll
    for (int ni = 0; ni < 4; ++ni) {
      const int j = ni * 16 + colL;
      const int yj = j >> 3, xj = j & 7;
      const int rj = region(wy * 8 + yj) * 3 + region(wx * 8 + xj);
      const int rel = (yi - yj + 7) * 15 + (xi - xj + 7);
      float s = acc[ni][r] + lsBias[rel] + ((ri != rj) ? -100.0f : 0.0f);
      vals[ni] = s;
      vmax = fmaxf(vmax, s);
    }
#pragma unroll
    for (int m = 1; m < 16; m <<= 1) vmax = fmaxf(vmax, __shfl_xor(vmax, m, 32));
    float ssum = 0.0f;
#pragma unroll
    for (int ni = 0; ni < 4; ++ni) { vals[ni] = __expf(vals[ni] - vmax); ssum += vals[ni]; }
#pragma unroll
    for (int m = 1; m < 16; m <<= 1) ssum += __shfl_xor(ssum, m, 32);
    const float inv = 1.0f / ssum;
#pragma unroll
    for (int ni = 0; ni < 4; ++ni)
      lsP[i * 72 + ni * 16 + colL] = (_Float16)(vals[ni] * inv);
  }

  // P @ V : each wave reads only its own 16 rows of P (no barrier needed)
  v8f accv[2];
  accv[0] = (v8f)0.0f; accv[1] = (v8f)0.0f;
#pragma unroll
  for (int kk = 0; kk < 2; ++kk) {
    v16h ap = ld_afrag(lsP, r0, kk * 32, 72, lane);
#pragma unroll
    for (int ni = 0; ni < 2; ++ni) {
      v16h bv = ld_bfrag(lsVt, ni * 16, kk * 32, 72, lane);
      accv[ni] = WMMA_F16(ap, bv, accv[ni]);
    }
  }
#pragma unroll
  for (int ni = 0; ni < 2; ++ni)
#pragma unroll
    for (int r = 0; r < 8; ++r) {
      const int i = r0 + r + rhalf * 8;
      out16[(size_t)(win * 64 + i) * kC + head * kHD + ni * 16 + colL] =
          (_Float16)accv[ni][r];
    }
}

// --------------------------- launch ----------------------------------------
extern "C" void kernel_launch(void* const* d_in, const int* in_sizes, int n_in,
                              void* d_out, int out_size, void* d_ws, size_t ws_size,
                              hipStream_t stream) {
  const float* x      = (const float*)d_in[0];
  const float* n1g    = (const float*)d_in[1];
  const float* n1b    = (const float*)d_in[2];
  const float* qkv_w  = (const float*)d_in[3];
  const float* qkv_b  = (const float*)d_in[4];
  const float* rpb    = (const float*)d_in[5];
  const float* proj_w = (const float*)d_in[6];
  const float* proj_b = (const float*)d_in[7];
  const float* n2g    = (const float*)d_in[8];
  const float* n2b    = (const float*)d_in[9];
  const float* fc1_w  = (const float*)d_in[10];
  const float* fc1_b  = (const float*)d_in[11];
  const float* fc2_w  = (const float*)d_in[12];
  const float* fc2_b  = (const float*)d_in[13];
  float* out = (float*)d_out;

  char* ws = (char*)d_ws;
  _Float16* buf0 = (_Float16*)(ws + OFF_BUF0);   // xw16 -> attn16 -> m16
  _Float16* q16  = (_Float16*)(ws + OFF_Q);
  _Float16* k16  = (_Float16*)(ws + OFF_K);
  _Float16* v16  = (_Float16*)(ws + OFF_V);
  _Float16* h16  = (_Float16*)(ws + OFF_H16);
  float*    x1   = (float*)(ws + OFF_X1);
  _Float16* wqkv = (_Float16*)(ws + OFF_WQKV);
  _Float16* wprj = (_Float16*)(ws + OFF_WPRJ);
  _Float16* wfc1 = (_Float16*)(ws + OFF_WFC1);
  _Float16* wfc2 = (_Float16*)(ws + OFF_WFC2);

  // fp16 weight copies (cheap: ~0.8M elements total)
  cvt_f32_f16<<<768, 256, 0, stream>>>(qkv_w, wqkv, 768 * 256);
  cvt_f32_f16<<<256, 256, 0, stream>>>(proj_w, wprj, 256 * 256);
  cvt_f32_f16<<<1024, 256, 0, stream>>>(fc1_w, wfc1, 1024 * 256);
  cvt_f32_f16<<<1024, 256, 0, stream>>>(fc2_w, wfc2, 256 * 1024);

  // LN1 + cyclic shift + window partition -> xw16 (windowed order)
  ln_kernel<0><<<kT / 8, 256, 0, stream>>>(x, n1g, n1b, buf0);

  // QKV: 131072 x 768 x 256 (blockIdx.y selects q/k/v plane)
  gemm_wmma_kernel<256, EPI_QKV><<<dim3(kT / 128, 3), 256, 0, stream>>>(
      buf0, wqkv, qkv_b, q16, k16, v16, nullptr, nullptr);

  // attention per (head, window); writes attn16 into buf0 (xw16 dead)
  attn_kernel<<<dim3(kNH, kB * kNW), 128, 0, stream>>>(q16, k16, v16, rpb, buf0);

  // proj + window reverse + shift + residual -> x1 (f32)
  gemm_wmma_kernel<256, EPI_PROJ><<<dim3(kT / 128, 1), 256, 0, stream>>>(
      buf0, wprj, proj_b, nullptr, nullptr, nullptr, x, x1);

  // LN2 -> m16 (natural order, buf0 reused)
  ln_kernel<1><<<kT / 8, 256, 0, stream>>>(x1, n2g, n2b, buf0);

  // FC1 + exact GELU -> h16 (reuses dead qkv region)
  gemm_wmma_kernel<256, EPI_FC1><<<dim3(kT / 128, 4), 256, 0, stream>>>(
      buf0, wfc1, fc1_b, h16, nullptr, nullptr, nullptr, nullptr);

  // FC2 + residual(x1) -> out (f32)
  gemm_wmma_kernel<1024, EPI_FC2><<<dim3(kT / 128, 1), 256, 0, stream>>>(
      h16, wfc2, fc2_b, nullptr, nullptr, nullptr, x1, out);
}
